// DBOT_Sinkhorn_Loss_57801669869627
// MI455X (gfx1250) — matrix operand order repaired and checked by reading, not compiled, not run on hardware
//
#include <hip/hip_runtime.h>
#include <hip/hip_bf16.h>
#include <cstdint>

// ---------------------------------------------------------------------------
// DBOT Sinkhorn loss, MI455X (gfx1250, wave32).
//   E = exp(img@txt^T - 1)  stored fp16 (128 MB -> resident in 192 MB L2)
//   Sinkhorn kept implicit: P1 = E*(r x c), P2 = E^T*(r2 x c2)
//   11 streaming passes over E total (10 sinkhorn + 1 CE), all L2-resident.
// ---------------------------------------------------------------------------

constexpr int N  = 8192;   // batch
constexpr int DK = 256;    // feature dim
constexpr int NSTRIPS = N / 64;  // 128 row strips for reduction passes

typedef _Float16 v16h __attribute__((ext_vector_type(16)));
typedef _Float16 h8   __attribute__((ext_vector_type(8)));
typedef float    v8f  __attribute__((ext_vector_type(8)));

// ---------------- small helpers -------------------------------------------

__global__ void k_init_vec(float* r, float* c, float* r2, float* c2) {
  int i = blockIdx.x * blockDim.x + threadIdx.x;
  if (i < N) { r[i] = 1.f; c[i] = 1.f; r2[i] = 1.f; c2[i] = 1.f; }
}

__global__ void k_cvt(const float* __restrict__ img, const float* __restrict__ txt,
                      _Float16* __restrict__ imgh, _Float16* __restrict__ txth) {
  int i = blockIdx.x * blockDim.x + threadIdx.x;
  if (i < N * DK) { imgh[i] = (_Float16)img[i]; txth[i] = (_Float16)txt[i]; }
}

// ---------------- WMMA GEMM + exp epilogue --------------------------------
// E[i][j] = exp(img_i . txt_j - 1), fp16 output.
// Wave computes a 16x64 strip: one A fragment, 4 B fragments / K-step.
// Fragment layouts per CDNA5 ISA 7.12.2 (16-bit A 16x32, B 32x16, f32 C/D).

__global__ __launch_bounds__(256)
void k_gemm_exp(const _Float16* __restrict__ A, const _Float16* __restrict__ B,
                _Float16* __restrict__ E) {
  const int wave = threadIdx.x >> 5;
  const int lane = threadIdx.x & 31;
  const int half = lane >> 4;     // lane group 0/1
  const int l16  = lane & 15;
  const int i0 = blockIdx.y * 16;                  // 512 row tiles
  const int j0 = (blockIdx.x * 8 + wave) * 64;     // 16 blocks * 8 waves * 64 cols

  const _Float16* Ap = A + (size_t)(i0 + l16) * DK;

  v8f acc[4];
  #pragma unroll
  for (int jj = 0; jj < 4; ++jj) acc[jj] = (v8f){0.f,0.f,0.f,0.f,0.f,0.f,0.f,0.f};

  #pragma unroll
  for (int kk = 0; kk < DK; kk += 32) {
    // A 16x32 f16 fragment: lane<16 -> row l16, halves = K[0..7]++K[16..23];
    // lane>=16 -> K[8..15]++K[24..31]
    h8 alo = *(const h8*)(Ap + kk + half * 8);
    h8 ahi = *(const h8*)(Ap + kk + 16 + half * 8);
    v16h a = __builtin_shufflevector(alo, ahi, 0,1,2,3,4,5,6,7,8,9,10,11,12,13,14,15);
    #pragma unroll
    for (int jj = 0; jj < 4; ++jj) {
      // B 32x16 fragment with Bfrag[k][n] = txt[j0+n][kk+k]:
      // lane<16 -> col n=l16, halves = K[0..7]++K[16..23];
      // lane>=16 -> K[8..15]++K[24..31]
      const _Float16* Bp = B + (size_t)(j0 + jj * 16 + l16) * DK;
      h8 blo = *(const h8*)(Bp + kk + half * 8);
      h8 bhi = *(const h8*)(Bp + kk + 16 + half * 8);
      v16h b = __builtin_shufflevector(blo, bhi, 0,1,2,3,4,5,6,7,8,9,10,11,12,13,14,15);
      acc[jj] = __builtin_amdgcn_wmma_f32_16x16x32_f16(
          false, a, false, b, (short)0, acc[jj], false, false);
    }
  }

  // C/D layout: VGPR e, lanes 0-15: M=e, N=lane; lanes 16-31: M=e+8, N=lane-16
  #pragma unroll
  for (int jj = 0; jj < 4; ++jj) {
    size_t base = (size_t)(i0 + half * 8) * N + (j0 + jj * 16 + l16);
    #pragma unroll
    for (int e = 0; e < 8; ++e)
      E[base + (size_t)e * N] = (_Float16)__expf(acc[jj][e] - 1.0f);
  }
}

// ---------------- fused weighted row+col reduction pass -------------------
// rowred[i]      = sum_j E[i][j] * cw[j]
// colpart[s][j]  = sum_{i in strip s} E[i][j] * rw[i]   (reduced later)

__global__ __launch_bounds__(256)
void k_pass(const _Float16* __restrict__ E, const float* __restrict__ cw,
            const float* __restrict__ rw, float* __restrict__ rowred,
            float* __restrict__ colpart) {
  __shared__ float rp[64];
  const int tid = threadIdx.x, wave = tid >> 5, lane = tid & 31;
  const int row0 = blockIdx.x * 64;
  if (tid < 64) rp[tid] = 0.f;
  __syncthreads();

  for (int g = wave; g < 32; g += 8) {          // 256-col groups
    const int j0 = g * 256 + lane * 8;          // 8 cols per lane
    float cwv[8], colacc[8];
    #pragma unroll
    for (int k = 0; k < 8; ++k) { cwv[k] = cw[j0 + k]; colacc[k] = 0.f; }
    for (int rr = 0; rr < 64; ++rr) {
      h8 v = *(const h8*)(E + (size_t)(row0 + rr) * N + j0);
      float rwv = rw[row0 + rr];
      float s = 0.f;
      #pragma unroll
      for (int k = 0; k < 8; ++k) {
        float f = (float)v[k];
        s += f * cwv[k];
        colacc[k] += f * rwv;
      }
      #pragma unroll
      for (int d = 16; d >= 1; d >>= 1) s += __shfl_xor(s, d, 32);
      if (lane == 0) atomicAdd(&rp[rr], s);     // ds_add_f32
    }
    #pragma unroll
    for (int k = 0; k < 8; ++k)
      colpart[(size_t)blockIdx.x * N + j0 + k] = colacc[k];
  }
  __syncthreads();
  if (tid < 64) rowred[row0 + tid] = rp[tid];
}

// ---------------- CE pass: row/col sum-exp of implicit P1/P2 --------------
// rowse[i]      = sum_j exp(E_ij * r_i * c_j)            (P1 row logits)
// colpart[s][j] = sum_{i in s} exp(E_ij * r2_j * c2_i)   (P2 row j logits)

__global__ __launch_bounds__(256)
void k_cepass(const _Float16* __restrict__ E, const float* __restrict__ r,
              const float* __restrict__ c, const float* __restrict__ r2,
              const float* __restrict__ c2, float* __restrict__ rowse,
              float* __restrict__ colpart) {
  __shared__ float rp[64];
  const int tid = threadIdx.x, wave = tid >> 5, lane = tid & 31;
  const int row0 = blockIdx.x * 64;
  if (tid < 64) rp[tid] = 0.f;
  __syncthreads();

  for (int g = wave; g < 32; g += 8) {
    const int j0 = g * 256 + lane * 8;
    float cv[8], r2v[8], colacc[8];
    #pragma unroll
    for (int k = 0; k < 8; ++k) { cv[k] = c[j0+k]; r2v[k] = r2[j0+k]; colacc[k] = 0.f; }
    for (int rr = 0; rr < 64; ++rr) {
      h8 v = *(const h8*)(E + (size_t)(row0 + rr) * N + j0);
      float ri  = r[row0 + rr];
      float c2i = c2[row0 + rr];
      float s = 0.f;
      #pragma unroll
      for (int k = 0; k < 8; ++k) {
        float f = (float)v[k];                 // values in (0,1]; exp args small
        s += __expf(f * ri * cv[k]);
        colacc[k] += __expf(f * r2v[k] * c2i);
      }
      #pragma unroll
      for (int d = 16; d >= 1; d >>= 1) s += __shfl_xor(s, d, 32);
      if (lane == 0) atomicAdd(&rp[rr], s);
    }
    #pragma unroll
    for (int k = 0; k < 8; ++k)
      colpart[(size_t)blockIdx.x * N + j0 + k] = colacc[k];
  }
  __syncthreads();
  if (tid < 64) rowse[row0 + tid] = rp[tid];
}

// ---------------- vector updates ------------------------------------------
// After pass A (cw=c, rw=c2): rowred=S1 -> r=1/S1 ; colpart=S2 -> r2=1/S2
__global__ void k_updA(const float* __restrict__ rowred, const float* __restrict__ colpart,
                       float* __restrict__ r, float* __restrict__ r2) {
  int i = blockIdx.x * blockDim.x + threadIdx.x;
  if (i >= N) return;
  r[i] = 1.0f / rowred[i];
  float s = 0.f;
  for (int b = 0; b < NSTRIPS; ++b) s += colpart[(size_t)b * N + i];
  r2[i] = 1.0f / s;
}

// After pass B (cw=r2, rw=r): rowred=T2 -> update c2 ; colpart=T1 -> update c
// clamp-up then clamp-down; second colsum is analytic (c*T), no extra pass.
__global__ void k_updB(const float* __restrict__ rowred, const float* __restrict__ colpart,
                       float* __restrict__ c2, float* __restrict__ c) {
  int i = blockIdx.x * blockDim.x + threadIdx.x;
  if (i >= N) return;
  const float bd = 0.5f, bu = 4.5f;            // 0.1*n, 0.9*n with n=5
  float T2 = rowred[i];
  float v = c2[i];
  v *= fmaxf(bd / (v * T2), 1.0f);
  v *= fminf(bu / (v * T2), 1.0f);
  c2[i] = v;
  float T1 = 0.f;
  for (int b = 0; b < NSTRIPS; ++b) T1 += colpart[(size_t)b * N + i];
  float w = c[i];
  w *= fmaxf(bd / (w * T1), 1.0f);
  w *= fminf(bu / (w * T1), 1.0f);
  c[i] = w;
}

// ---------------- final CE reduction --------------------------------------
__global__ __launch_bounds__(256)
void k_fin1(const _Float16* __restrict__ E, const float* __restrict__ rowse,
            const float* __restrict__ colpart, const float* __restrict__ r,
            const float* __restrict__ c, const float* __restrict__ r2,
            const float* __restrict__ c2, const int* __restrict__ labels,
            float* __restrict__ part) {
  __shared__ float red[256];
  int i = blockIdx.x * 256 + threadIdx.x;
  float se1 = rowse[i];
  float se2 = 0.f;
  for (int b = 0; b < NSTRIPS; ++b) se2 += colpart[(size_t)b * N + i];
  int l = labels[i];
  float p1 = (float)E[(size_t)i * N + l] * r[i]  * c[l];    // P1[i, l]
  float p2 = (float)E[(size_t)l * N + i] * r2[i] * c2[l];   // P2[i, l] = E[l][i]*r2_i*c2_l
  float t = (__logf(se1) - p1) + (__logf(se2) - p2);
  red[threadIdx.x] = t;
  __syncthreads();
  for (int s = 128; s > 0; s >>= 1) {
    if (threadIdx.x < s) red[threadIdx.x] += red[threadIdx.x + s];
    __syncthreads();
  }
  if (threadIdx.x == 0) part[blockIdx.x] = red[0];
}

__global__ void k_fin2(const float* __restrict__ part, float* __restrict__ out) {
  float s = part[threadIdx.x];                 // 32 partials, one wave
  #pragma unroll
  for (int d = 16; d >= 1; d >>= 1) s += __shfl_xor(s, d, 32);
  if (threadIdx.x == 0) out[0] = s / (2.0f * (float)N);
}

// ---------------- launch ---------------------------------------------------
extern "C" void kernel_launch(void* const* d_in, const int* in_sizes, int n_in,
                              void* d_out, int out_size, void* d_ws, size_t ws_size,
                              hipStream_t stream) {
  const float* img    = (const float*)d_in[0];
  const float* txt    = (const float*)d_in[1];
  const int*   labels = (const int*)d_in[2];
  float* out = (float*)d_out;

  // workspace layout (all 512B-aligned); total ~140.2 MiB
  char* ws = (char*)d_ws;
  size_t off = 0;
  _Float16* E    = (_Float16*)(ws + off); off += (size_t)N * N * sizeof(_Float16);   // 128 MiB
  _Float16* imgh = (_Float16*)(ws + off); off += (size_t)N * DK * sizeof(_Float16);
  _Float16* txth = (_Float16*)(ws + off); off += (size_t)N * DK * sizeof(_Float16);
  float* rv      = (float*)(ws + off);    off += (size_t)N * sizeof(float);
  float* cv      = (float*)(ws + off);    off += (size_t)N * sizeof(float);
  float* r2v     = (float*)(ws + off);    off += (size_t)N * sizeof(float);
  float* c2v     = (float*)(ws + off);    off += (size_t)N * sizeof(float);
  float* rowred  = (float*)(ws + off);    off += (size_t)N * sizeof(float);
  float* colpart = (float*)(ws + off);    off += (size_t)NSTRIPS * N * sizeof(float); // 4 MiB
  float* part    = (float*)(ws + off);    off += 512;

  k_init_vec<<<32, 256, 0, stream>>>(rv, cv, r2v, c2v);
  k_cvt<<<(N * DK + 255) / 256, 256, 0, stream>>>(img, txt, imgh, txth);
  k_gemm_exp<<<dim3(16, 512), 256, 0, stream>>>(imgh, txth, E);

  for (int it = 0; it < 5; ++it) {
    // pass A: rowred = S1 (weights c), colpart -> S2 (weights c2)
    k_pass<<<NSTRIPS, 256, 0, stream>>>(E, cv, c2v, rowred, colpart);
    k_updA<<<32, 256, 0, stream>>>(rowred, colpart, rv, r2v);
    // pass B: rowred = T2 (weights r2), colpart -> T1 (weights r)
    k_pass<<<NSTRIPS, 256, 0, stream>>>(E, r2v, rv, rowred, colpart);
    k_updB<<<32, 256, 0, stream>>>(rowred, colpart, c2v, cv);
  }

  k_cepass<<<NSTRIPS, 256, 0, stream>>>(E, rv, cv, r2v, c2v, rowred, colpart);
  k_fin1<<<32, 256, 0, stream>>>(E, rowred, colpart, rv, cv, r2v, c2v, labels, part);
  k_fin2<<<1, 32, 0, stream>>>(part, out);
}
